// FastFoveatedConv2d_7997229105539
// MI455X (gfx1250) — matrix operand order repaired and checked by reading, not compiled, not run on hardware
//
#include <hip/hip_runtime.h>

typedef __attribute__((ext_vector_type(16))) _Float16 v16h;
typedef __attribute__((ext_vector_type(8)))  _Float16 v8h;
typedef __attribute__((ext_vector_type(8)))  float    v8f;

// ---------------------------------------------------------------------------
// Compile-time replica of the Python sampling-coordinate generator, reduced to
// axis-aligned run descriptors (every position is <= 2 runs: vertical and/or
// horizontal), evaluated against row/col prefix sums.
// ---------------------------------------------------------------------------
struct Run { signed char axis, fix, lo, hi; };   // axis: 0=horiz (vary dw), 1=vert (vary dh)
struct Tables {
    Run runs[49][2];
    signed char nruns[49];
    float invn[49];
};

constexpr void emitPos(Tables& t, int& pi, const int (&dh)[40], const int (&dw)[40], int n) {
    int nr = 0, s = 0;
    while (s < n) {
        int e = s + 1;
        if (e < n && dw[e] == dw[s]) {          // vertical run (constant dw)
            while (e < n && dw[e] == dw[s]) ++e;
            int lo = dh[s], hi = dh[s];
            for (int q = s; q < e; ++q) { if (dh[q] < lo) lo = dh[q]; if (dh[q] > hi) hi = dh[q]; }
            t.runs[pi][nr] = Run{(signed char)1, (signed char)dw[s], (signed char)lo, (signed char)hi};
        } else {                                 // horizontal run (constant dh), covers singles
            while (e < n && dh[e] == dh[s]) ++e;
            int lo = dw[s], hi = dw[s];
            for (int q = s; q < e; ++q) { if (dw[q] < lo) lo = dw[q]; if (dw[q] > hi) hi = dw[q]; }
            t.runs[pi][nr] = Run{(signed char)0, (signed char)dh[s], (signed char)lo, (signed char)hi};
        }
        ++nr; s = e;
    }
    t.nruns[pi] = (signed char)nr;
    t.invn[pi]  = 1.0f / (float)n;
    ++pi;
}

constexpr Tables makeTables() {
    Tables t{};
    int pi = 0;
    int dh[40] = {}, dw[40] = {};
    // inner 3x3 singles
    for (int i = -1; i <= 1; ++i)
        for (int j = -1; j <= 1; ++j) { dh[0] = i; dw[0] = j; emitPos(t, pi, dh, dw, 1); }
    // ring5
    for (int i = -2; i <= 2; ++i)
        for (int j = -2; j <= 2; ++j) {
            if (!(i == -2 || i == 2 || j == -2 || j == 2)) continue;
            int n = 0;
            if (i == -2 && j >= -1 && j <= 1)      { for (int k = 0; k < 5; ++k) { dh[n] = i + k; dw[n] = j; ++n; } }
            else if (i == 2 && j >= -1 && j <= 1)  { for (int k = 0; k < 5; ++k) { dh[n] = i - k; dw[n] = j; ++n; } }
            else if (j == -2 && i >= -1 && i <= 1) { for (int k = 0; k < 5; ++k) { dh[n] = i; dw[n] = j + k; ++n; } }
            else if (j == 2 && i >= -1 && i <= 1)  { for (int k = 0; k < 5; ++k) { dh[n] = i; dw[n] = j - k; ++n; } }
            else if (i == -2 && j == -2) { for (int k = 0; k < 5; ++k) { dh[n] = i + k; dw[n] = j; ++n; } for (int k = 1; k < 5; ++k) { dh[n] = i; dw[n] = j + k; ++n; } }
            else if (i == -2 && j == 2)  { for (int k = 0; k < 5; ++k) { dh[n] = i + k; dw[n] = j; ++n; } for (int k = 1; k < 5; ++k) { dh[n] = i; dw[n] = j - k; ++n; } }
            else if (i == 2 && j == -2)  { for (int k = 0; k < 5; ++k) { dh[n] = i - k; dw[n] = j; ++n; } for (int k = 1; k < 5; ++k) { dh[n] = i; dw[n] = j + k; ++n; } }
            else                          { for (int k = 0; k < 5; ++k) { dh[n] = i - k; dw[n] = j; ++n; } for (int k = 1; k < 5; ++k) { dh[n] = i; dw[n] = j - k; ++n; } }
            emitPos(t, pi, dh, dw, n);
        }
    // ring7
    for (int i = -3; i <= 3; ++i)
        for (int j = -3; j <= 3; ++j) {
            if (!(i == -3 || i == 3 || j == -3 || j == 3)) continue;
            int n = 0;
            if (i == -3 && j >= -2 && j <= 2)      { for (int k = 0; k < 15; ++k) { dh[n] = i + k; dw[n] = j; ++n; } }
            else if (i == 3 && j >= -2 && j <= 2)  { for (int k = 0; k < 15; ++k) { dh[n] = i - k; dw[n] = j; ++n; } }
            else if (j == -3 && i >= -2 && i <= 2) { for (int k = 0; k < 25; ++k) { dh[n] = i; dw[n] = j + k; ++n; } }
            else if (j == 3 && i >= -2 && i <= 2)  { for (int k = 0; k < 25; ++k) { dh[n] = i; dw[n] = j - k; ++n; } }
            else if (i == -3 && j == -3) { for (int k = 0; k < 15; ++k) { dh[n] = i + k; dw[n] = j; ++n; } for (int k = 1; k < 25; ++k) { dh[n] = i; dw[n] = j + k; ++n; } }
            else if (i == -3 && j == 3)  { for (int k = 0; k < 15; ++k) { dh[n] = i + k; dw[n] = j; ++n; } for (int k = 1; k < 25; ++k) { dh[n] = i; dw[n] = j - k; ++n; } }
            else if (i == 3 && j == -3)  { for (int k = 0; k < 15; ++k) { dh[n] = i - k; dw[n] = j; ++n; } for (int k = 1; k < 25; ++k) { dh[n] = i; dw[n] = j + k; ++n; } }
            else                          { for (int k = 0; k < 15; ++k) { dh[n] = i - k; dw[n] = j; ++n; } for (int k = 1; k < 25; ++k) { dh[n] = i; dw[n] = j - k; ++n; } }
            emitPos(t, pi, dh, dw, n);
        }
    return t;
}

__constant__ Tables g_tab = makeTables();

// ---------------------------------------------------------------------------
// Fused foveated-feature + (M=128, N=64, K=32*64) WMMA GEMM kernel.
// Block: 256 threads = 8 wave32; tile = 8 rows x 16 cols of output pixels.
// __launch_bounds__(256, 8): keep >= 8 waves/SIMD (>= 4 blocks/WGP) so the
// serial prefix-scan phase and barriers are hidden by block concurrency.
// ---------------------------------------------------------------------------
#define XSTR 60   // f32 stride for 30x58 tile / prefix arrays
#define FSTR 72   // f16 stride (144B = 36 dwords, bank-friendly)
#define BSTR 72

__global__ __launch_bounds__(256, 8) void fovconv_kernel(
    const float* __restrict__ x, const float* __restrict__ wgt,
    const float* __restrict__ bias, float* __restrict__ out)
{
    __shared__ float    xt [30 * XSTR];   // clamped/reflected padded window
    __shared__ float    RPa[30 * XSTR];   // row prefix sums  (width 59 used)
    __shared__ float    CPa[31 * XSTR];   // col prefix sums  (height 31 used)
    __shared__ _Float16 F  [128 * FSTR];  // per-channel feature strip, K padded 49->64
    __shared__ _Float16 Bw [64 * BSTR];   // per-channel weights W[o][p], padded

    const int tid  = threadIdx.x;
    const int lane = tid & 31;
    const int wave = tid >> 5;
    const int b    = blockIdx.z;
    const int h0   = blockIdx.y * 8;
    const int w0   = blockIdx.x * 16;

    // Zero the K-padding (p = 49..63) once; never rewritten inside the loop.
    if (tid < 128) {
        #pragma unroll
        for (int k = 49; k < 64; ++k) F[tid * FSTR + k] = (_Float16)0.0f;
    }
    if (tid < 64) {
        #pragma unroll
        for (int k = 49; k < 64; ++k) Bw[tid * BSTR + k] = (_Float16)0.0f;
    }

    v8f acc[4] = {};   // 16 pixels x 64 outputs, f32 accumulation

    const float* xplane = x + ((long)b << 21);   // b * 32 * 65536

    for (int c = 0; c < 32; ++c) {
        const float* xp = xplane + (c << 16);

        // ---- stage x window: padded rows [h0-8, h0+21], cols [w0-18, w0+39],
        //      values resolved through clamp(padded bound) + reflect(pad=3).
        //      Two-phase: batch all global loads into registers (loads pipeline
        //      behind ONE loadcnt wait), then drain to LDS.
        {
            float v[7];
            int   ofs[7];
            #pragma unroll
            for (int u = 0; u < 7; ++u) {
                int t = tid + (u << 8);
                if (t < 30 * 58) {
                    int i = t / 58, j = t - i * 58;
                    int pr = h0 - 8 + i;  pr = pr < 0 ? 0 : (pr > 261 ? 261 : pr);
                    int pc = w0 - 18 + j; pc = pc < 0 ? 0 : (pc > 261 ? 261 : pc);
                    int xr = pr - 3; xr = xr < 0 ? -xr : (xr > 255 ? 510 - xr : xr);
                    int xw = pc - 3; xw = xw < 0 ? -xw : (xw > 255 ? 510 - xw : xw);
                    v[u]   = xp[(xr << 8) + xw];
                    ofs[u] = i * XSTR + j;
                }
            }
            #pragma unroll
            for (int u = 0; u < 7; ++u) {
                int t = tid + (u << 8);
                if (t < 30 * 58) xt[ofs[u]] = v[u];
            }
        }
        // ---- stage channel-c weights as f16: Bw[o][p] = W[o,c,p] (same
        //      two-phase batching; 3136 = 12*256 + 64)
        {
            const float* wp = wgt + c * 49;
            float wv[13];
            #pragma unroll
            for (int u = 0; u < 13; ++u) {
                int t = tid + (u << 8);
                if (t < 64 * 49) {
                    int o = t / 49, p = t - o * 49;
                    wv[u] = wp[o * (32 * 49) + p];
                }
            }
            #pragma unroll
            for (int u = 0; u < 13; ++u) {
                int t = tid + (u << 8);
                if (t < 64 * 49) {
                    int o = t / 49, p = t - o * 49;
                    Bw[o * BSTR + p] = (_Float16)wv[u];
                }
            }
        }
        // ---- software-pipeline: prefetch next channel's window into the WGP
        //      (two 128B lines per row cover all 58 floats; addresses are the
        //      clamp/reflect-resolved ones, so they are always valid).
        if (c + 1 < 32) {
            const float* xn = xp + (1 << 16);
            for (int t = tid; t < 60; t += 256) {
                int i = t >> 1, j = (t & 1) * 29;
                int pr = h0 - 8 + i;  pr = pr < 0 ? 0 : (pr > 261 ? 261 : pr);
                int pc = w0 - 18 + j; pc = pc < 0 ? 0 : (pc > 261 ? 261 : pc);
                int xr = pr - 3; xr = xr < 0 ? -xr : (xr > 255 ? 510 - xr : xr);
                int xw = pc - 3; xw = xw < 0 ? -xw : (xw > 255 ? 510 - xw : xw);
                __builtin_prefetch(xn + (xr << 8) + xw, 0, 3);
            }
        }
        __syncthreads();

        // ---- prefix sums (rows by threads 0..29, cols by threads 32..89)
        if (tid < 30) {
            int base = tid * XSTR;
            float s = 0.0f;
            RPa[base] = 0.0f;
            for (int j = 0; j < 58; ++j) { s += xt[base + j]; RPa[base + j + 1] = s; }
        } else if (tid >= 32 && tid < 32 + 58) {
            int col = tid - 32;
            float s = 0.0f;
            CPa[col] = 0.0f;
            for (int i = 0; i < 30; ++i) { s += xt[i * XSTR + col]; CPa[(i + 1) * XSTR + col] = s; }
        }
        __syncthreads();

        // ---- 49 features per pixel, each <= 2 segment differences
        //      (p = t>>7 is wave-uniform -> table reads scalarize)
        for (int t = tid; t < 49 * 128; t += 256) {
            int p = t >> 7, pix = t & 127;
            int r  = (pix >> 4) + 11;     // local padded-row of pixel center
            int cc = (pix & 15) + 21;     // local padded-col of pixel center
            float a = 0.0f;
            int nr = g_tab.nruns[p];
            #pragma unroll
            for (int q = 0; q < 2; ++q) {
                if (q < nr) {
                    Run rn = g_tab.runs[p][q];
                    if (rn.axis) {
                        int col = cc + rn.fix;
                        a += CPa[(r + rn.hi + 1) * XSTR + col] - CPa[(r + rn.lo) * XSTR + col];
                    } else {
                        int row = r + rn.fix;
                        a += RPa[row * XSTR + cc + rn.hi + 1] - RPa[row * XSTR + cc + rn.lo];
                    }
                }
            }
            F[pix * FSTR + p] = (_Float16)(a * g_tab.invn[p]);
        }
        __syncthreads();

        // ---- WMMA: per K-chunk, front-load A + all 4 B fragments (10 b128
        //      LDS loads in flight), then issue 4 back-to-back WMMAs on
        //      independent accumulators (no D->A/B hazard, partial dscnt waits).
        {
            const int m = lane & 15, g = lane >> 4;
            const _Float16* Fp = &F[((wave << 4) + m) * FSTR];
            #pragma unroll
            for (int kc = 0; kc < 2; ++kc) {
                const int kb = kc << 5;
                // A 16x32 f16: lane half g needs K = {kb+8g..+7} and {kb+16+8g..+7}
                v8h alo = *(const v8h*)(Fp + kb + (g << 3));
                v8h ahi = *(const v8h*)(Fp + kb + 16 + (g << 3));
                v16h a;
                #pragma unroll
                for (int e = 0; e < 8; ++e) { a[e] = alo[e]; a[e + 8] = ahi[e]; }
                v16h bfr[4];
                #pragma unroll
                for (int nt = 0; nt < 4; ++nt) {
                    // B 32x16 f16: lane = column n, half g holds K = {kb+8g..+7} and {kb+16+8g..+7}
                    const _Float16* Bp = &Bw[((nt << 4) + m) * BSTR + kb + (g << 3)];
                    v8h blo = *(const v8h*)(Bp);
                    v8h bhi = *(const v8h*)(Bp + 16);
                    #pragma unroll
                    for (int e = 0; e < 8; ++e) { bfr[nt][e] = blo[e]; bfr[nt][e + 8] = bhi[e]; }
                }
                #pragma unroll
                for (int nt = 0; nt < 4; ++nt) {
                    acc[nt] = __builtin_amdgcn_wmma_f32_16x16x32_f16(
                        false, a, false, bfr[nt], (short)0, acc[nt], false, false);
                }
            }
        }
        __syncthreads();
    }

    // ---- epilogue: D layout (m = vgpr + 8*(lane>=16), n = lane%16) -> the 8
    //      accumulator floats are consecutive in w: two float4 stores + bias.
    const int m16 = lane & 15, g = lane >> 4;
    const int h = h0 + wave;
    #pragma unroll
    for (int nt = 0; nt < 4; ++nt) {
        int o = (nt << 4) + m16;
        float bv = bias[o];
        float* op = out + (((((b << 6) + o) << 8) + h) << 8) + w0 + (g << 3);
        float4 v0 = make_float4(acc[nt][0] + bv, acc[nt][1] + bv, acc[nt][2] + bv, acc[nt][3] + bv);
        float4 v1 = make_float4(acc[nt][4] + bv, acc[nt][5] + bv, acc[nt][6] + bv, acc[nt][7] + bv);
        *(float4*)op       = v0;
        *(float4*)(op + 4) = v1;
    }
}

extern "C" void kernel_launch(void* const* d_in, const int* in_sizes, int n_in,
                              void* d_out, int out_size, void* d_ws, size_t ws_size,
                              hipStream_t stream) {
    (void)in_sizes; (void)n_in; (void)out_size; (void)d_ws; (void)ws_size;
    const float* x  = (const float*)d_in[0];   // (4, 32, 256, 256) f32
    const float* W  = (const float*)d_in[1];   // (64, 32, 49) f32
    const float* bb = (const float*)d_in[2];   // (64,) f32
    float* out = (float*)d_out;                // (4, 64, 256, 256) f32

    dim3 grid(256 / 16, 256 / 8, 4);           // (w-tiles, h-tiles, batch)
    dim3 block(256);
    fovconv_kernel<<<grid, block, 0, stream>>>(x, W, bb, out);
}
